// MultiheadAttention_14310831030543
// MI455X (gfx1250) — matrix operand order repaired and checked by reading, not compiled, not run on hardware
//
#include <hip/hip_runtime.h>
#include <hip/hip_bf16.h>
#include <math.h>

#define D_MODEL 1024
#define N_HEADS 16
#define HEAD_DIM 64
#define SEQ 2048
#define BATCH 4
#define M_TOTAL (BATCH * SEQ) /* 8192 */

typedef __attribute__((ext_vector_type(16))) __bf16 v16bf;
typedef __attribute__((ext_vector_type(8)))  __bf16 v8bf;
typedef __attribute__((ext_vector_type(8)))  float  v8f;
typedef int v4i __attribute__((vector_size(16)));   // matches builtin's GCC-vector param

// ---------------- CDNA5 async global->LDS copy (guarded) ----------------
#if defined(__AMDGCN__) && __has_builtin(__builtin_amdgcn_global_load_async_to_lds_b128)
#define HAVE_ASYNC_LDS 1
#else
#define HAVE_ASYNC_LDS 0
#endif

#if defined(__AMDGCN__) && __has_builtin(__builtin_amdgcn_s_wait_asynccnt)
#define WAIT_ASYNC(n) __builtin_amdgcn_s_wait_asynccnt(n)
#elif HAVE_ASYNC_LDS
#define WAIT_ASYNC(n) asm volatile("s_wait_asynccnt " #n ::: "memory")
#else
#define WAIT_ASYNC(n) ((void)0)
#endif

__device__ __forceinline__ void async_copy16(const __bf16* g, __bf16* l) {
#if HAVE_ASYNC_LDS
    __builtin_amdgcn_global_load_async_to_lds_b128(
        (v4i*)(const void*)g,   // global source (generic pointer per ROCm signature)
        (v4i*)(void*)l,         // LDS destination
        0, 0);
#else
    *(v8bf*)l = *(const v8bf*)g;   // synchronous fallback; __syncthreads provides visibility
#endif
}

__device__ __forceinline__ __bf16 f2bf(float f) {
    unsigned u = __builtin_bit_cast(unsigned, f);
    unsigned r = u + 0x7FFFu + ((u >> 16) & 1u);   // round-to-nearest-even
    unsigned short h = (unsigned short)(r >> 16);
    return __builtin_bit_cast(__bf16, h);
}

// ---- WMMA fragment loaders (CDNA5 16-bit layouts, wave32) ----
// A 16x32: lane<16 -> row=lane, elems[0..7]=K0..7, [8..15]=K16..23
//          lane>=16 -> row=lane-16, elems[0..7]=K8..15, [8..15]=K24..31
__device__ __forceinline__ v16bf load_a_frag(const __bf16* src, int ld, int lane) {
    int row = lane & 15, half = lane >> 4;
    const __bf16* p = src + (size_t)row * ld + half * 8;
    v8bf lo = *(const v8bf*)(p);
    v8bf hi = *(const v8bf*)(p + 16);
    v16bf a;
#pragma unroll
    for (int i = 0; i < 8; ++i) { a[i] = lo[i]; a[i + 8] = hi[i]; }
    return a;
}

// B 32x16 (src is column-contiguous, i.e. [N][K] row-major):
// lane<16 -> col=lane, elems[0..7]=K0..7, [8..15]=K16..23 ; lane>=16 -> col=lane-16, elems[0..7]=K8..15, [8..15]=K24..31
__device__ __forceinline__ v16bf load_b_frag(const __bf16* src, int ld, int lane) {
    int col = lane & 15, half = lane >> 4;
    const __bf16* p = src + (size_t)col * ld + half * 8;
    v8bf lo = *(const v8bf*)(p);
    v8bf hi = *(const v8bf*)(p + 16);
    v16bf b;
#pragma unroll
    for (int i = 0; i < 8; ++i) { b[i] = lo[i]; b[i + 8] = hi[i]; }
    return b;
}

__device__ __forceinline__ v8f wmma_bf16(v16bf a, v16bf b, v8f c) {
    return __builtin_amdgcn_wmma_f32_16x16x32_bf16(
        false, a, false, b, (short)0, c, false, false);
}

// ---------------- conversion kernels ----------------
__global__ void convert_x_kernel(const float* __restrict__ x, __bf16* __restrict__ xb, int n) {
    int i = (blockIdx.x * blockDim.x + threadIdx.x) * 8;
    if (i + 8 <= n) {
#pragma unroll
        for (int j = 0; j < 8; ++j) xb[i + j] = f2bf(x[i + j]);
    }
}

// W [K][N] fp32 row-major -> Wt [N][K] bf16 row-major (coalesced writes)
__global__ void transpose_w_kernel(const float* __restrict__ w, __bf16* __restrict__ wt) {
    int idx = blockIdx.x * blockDim.x + threadIdx.x;       // 0..1M
    int n = idx >> 10, k = idx & 1023;
    wt[idx] = f2bf(w[(size_t)k * D_MODEL + n]);
}

// ---------------- block-tiled GEMM: C[8192,1024] = A @ Wt^T + bias ----------------
// Block: 128(M) x 128(N), 8 waves (4Mx2N), wave: 32x64, K-step 32, double-buffered LDS
// mode: 0 = bf16 [B,H,S,Hd]; 1 = bf16 [B,H,Hd,S]; 2 = f32 row-major
#define GEMM_BK 32
__global__ __launch_bounds__(256) void gemm_wmma_kernel(
        const __bf16* __restrict__ A, const __bf16* __restrict__ Bt,
        const float* __restrict__ bias, void* __restrict__ out, int mode) {
    __shared__ __align__(16) __bf16 lA[2][128 * GEMM_BK];   // [m][k] 8KB each
    __shared__ __align__(16) __bf16 lB[2][128 * GEMM_BK];   // [n][k] 8KB each
    int tid = threadIdx.x;
    int wave = tid >> 5, lane = tid & 31;
    const int NBLK = D_MODEL / 128;                         // 8
    int bm = blockIdx.x / NBLK, bn = blockIdx.x % NBLK;
    const __bf16* Ab = A  + (size_t)bm * 128 * D_MODEL;
    const __bf16* Bb = Bt + (size_t)bn * 128 * D_MODEL;

    int crow = tid >> 2, cseg = (tid & 3) * 8;              // copy coords: 64 rows/group, 8-elem segs
    int wm = (wave >> 1) * 32, wn = (wave & 1) * 64;        // wave sub-tile origin

    v8f acc[2][4] = {};

    // stage k-slice into buffer
    auto stage = [&](int buf, int k0) {
        async_copy16(Ab + (size_t)(crow     ) * D_MODEL + k0 + cseg, &lA[buf][(crow     ) * GEMM_BK + cseg]);
        async_copy16(Ab + (size_t)(crow + 64) * D_MODEL + k0 + cseg, &lA[buf][(crow + 64) * GEMM_BK + cseg]);
        async_copy16(Bb + (size_t)(crow     ) * D_MODEL + k0 + cseg, &lB[buf][(crow     ) * GEMM_BK + cseg]);
        async_copy16(Bb + (size_t)(crow + 64) * D_MODEL + k0 + cseg, &lB[buf][(crow + 64) * GEMM_BK + cseg]);
    };

    stage(0, 0);
    int buf = 0;
    for (int k0 = 0; k0 < D_MODEL; k0 += GEMM_BK, buf ^= 1) {
        if (k0 + GEMM_BK < D_MODEL) {
            stage(buf ^ 1, k0 + GEMM_BK);   // prefetch next slice
            WAIT_ASYNC(4);                  // only the 4 just-issued may remain in flight
        } else {
            WAIT_ASYNC(0);
        }
        __syncthreads();
#pragma unroll
        for (int i = 0; i < 2; ++i) {
            v16bf a = load_a_frag(&lA[buf][(wm + i * 16) * GEMM_BK], GEMM_BK, lane);
#pragma unroll
            for (int t = 0; t < 4; ++t) {
                v16bf b = load_b_frag(&lB[buf][(wn + t * 16) * GEMM_BK], GEMM_BK, lane);
                acc[i][t] = wmma_bf16(a, b, acc[i][t]);
            }
        }
        __syncthreads();                    // everyone done reading buf before it is re-staged
    }

    int colL = lane & 15, halfL = lane >> 4;
#pragma unroll
    for (int i = 0; i < 2; ++i) {
#pragma unroll
        for (int t = 0; t < 4; ++t) {
            int gn = bn * 128 + wn + t * 16 + colL;
            float bval = bias[gn];
            int h = gn >> 6, hd = gn & 63;
#pragma unroll
            for (int r = 0; r < 8; ++r) {
                int gm = bm * 128 + wm + i * 16 + r + 8 * halfL;
                int bb = gm >> 11, s = gm & (SEQ - 1);
                float v = acc[i][t][r] + bval;
                if (mode == 0) {
                    ((__bf16*)out)[(((size_t)(bb * N_HEADS + h) * SEQ) + s) * HEAD_DIM + hd] = f2bf(v);
                } else if (mode == 1) {
                    ((__bf16*)out)[(((size_t)(bb * N_HEADS + h) * HEAD_DIM) + hd) * SEQ + s] = f2bf(v);
                } else {
                    ((float*)out)[(size_t)gm * D_MODEL + gn] = v;
                }
            }
        }
    }
}

// ---------------- block-cooperative causal flash attention ----------------
// Block = one (b,h) x 8 consecutive 16-query blocks; K/V tiles async-staged to LDS,
// double-buffered, shared by all 8 waves. Q,K: [B,H,S,Hd]; Vt: [B,H,Hd,S]; O: [B,S,D] bf16.
__global__ __launch_bounds__(256) void flash_attn_kernel(
        const __bf16* __restrict__ Q, const __bf16* __restrict__ K,
        const __bf16* __restrict__ Vt, __bf16* __restrict__ O) {
    __shared__ __align__(16) __bf16 lK[2][32 * 64];   // [key][hd]  4KB each
    __shared__ __align__(16) __bf16 lV[2][64 * 32];   // [hd][key]  4KB each
    __shared__ __align__(16) __bf16 lP[8][16 * 32];   // per-wave P tile
    int tid = threadIdx.x, wave = tid >> 5, lane = tid & 31;
    int bh = blockIdx.x >> 4;          // 0..63
    int qg = blockIdx.x & 15;          // q-group of 8 blocks
    int qb = qg * 8 + wave;            // this wave's 16-query block (0..127)

    const __bf16* Qm = Q  + ((size_t)bh * SEQ + qb * 16) * HEAD_DIM;
    const __bf16* Km = K  + (size_t)bh * SEQ * HEAD_DIM;
    const __bf16* Vm = Vt + (size_t)bh * HEAD_DIM * SEQ;

    v16bf qa0 = load_a_frag(Qm + 0,  HEAD_DIM, lane);
    v16bf qa1 = load_a_frag(Qm + 32, HEAD_DIM, lane);

    v8f o[4] = {};
    float mrow[8], lrow[8];
#pragma unroll
    for (int r = 0; r < 8; ++r) { mrow[r] = -INFINITY; lrow[r] = 0.f; }

    int halfL = lane >> 4, colL = lane & 15;
    int nkeys_w   = (qb + 1) * 16;
    int nkeys_max = (qg * 8 + 8) * 16;
    const float sc = 0.125f;           // 1/sqrt(64)

    int krow = tid >> 3, kseg = (tid & 7) * 8;  // K tile copy coords (32 x 64)
    int vrow = tid >> 2, vseg = (tid & 3) * 8;  // V tile copy coords (64 x 32)
    auto stage = [&](int buf, int j0) {
        async_copy16(Km + (size_t)(j0 + krow) * HEAD_DIM + kseg, &lK[buf][krow * 64 + kseg]);
        async_copy16(Vm + (size_t)vrow * SEQ + j0 + vseg,        &lV[buf][vrow * 32 + vseg]);
    };

    stage(0, 0);
    int buf = 0;
    for (int j0 = 0; j0 < nkeys_max; j0 += 32, buf ^= 1) {
        if (j0 + 32 < nkeys_max) {
            stage(buf ^ 1, j0 + 32);
            WAIT_ASYNC(2);
        } else {
            WAIT_ASYNC(0);
        }
        __syncthreads();

        if (j0 < nkeys_w) {                        // wave-uniform causal skip
            v8f s0 = {}, s1 = {};
            {
                v16bf b0 = load_b_frag(&lK[buf][0],  64, lane); s0 = wmma_bf16(qa0, b0, s0);
                v16bf b1 = load_b_frag(&lK[buf][32], 64, lane); s0 = wmma_bf16(qa1, b1, s0);
            }
            bool t1v = (j0 + 16) < nkeys_w;
            if (t1v) {
                v16bf b0 = load_b_frag(&lK[buf][16 * 64],      64, lane); s1 = wmma_bf16(qa0, b0, s1);
                v16bf b1 = load_b_frag(&lK[buf][16 * 64 + 32], 64, lane); s1 = wmma_bf16(qa1, b1, s1);
            }

            float p0[8], p1[8], alpha[8];
#pragma unroll
            for (int r = 0; r < 8; ++r) {
                int qi = qb * 16 + r + 8 * halfL;
                p0[r] = (j0 + colL <= qi) ? s0[r] * sc : -INFINITY;
                p1[r] = (t1v && (j0 + 16 + colL <= qi)) ? s1[r] * sc : -INFINITY;
            }
#pragma unroll
            for (int r = 0; r < 8; ++r) {
                float m = fmaxf(p0[r], p1[r]);
#pragma unroll
                for (int d = 1; d < 16; d <<= 1) m = fmaxf(m, __shfl_xor(m, d, 32));
                float mn = fmaxf(mrow[r], m);      // finite: diagonal block always present
                alpha[r] = __expf(mrow[r] - mn);
                mrow[r] = mn;
                p0[r] = __expf(p0[r] - mn);
                p1[r] = __expf(p1[r] - mn);
                float rs = p0[r] + p1[r];
#pragma unroll
                for (int d = 1; d < 16; d <<= 1) rs += __shfl_xor(rs, d, 32);
                lrow[r] = lrow[r] * alpha[r] + rs;
            }
#pragma unroll
            for (int t = 0; t < 4; ++t)
#pragma unroll
                for (int r = 0; r < 8; ++r) o[t][r] *= alpha[r];

            // P (C layout) -> wave-private LDS -> A layout
            __bf16* pl = &lP[wave][0];
#pragma unroll
            for (int r = 0; r < 8; ++r) {
                int row = r + 8 * halfL;
                pl[row * 32 + colL]      = f2bf(p0[r]);
                pl[row * 32 + 16 + colL] = f2bf(p1[r]);
            }
            asm volatile("s_wait_dscnt 0" ::: "memory");
            v16bf pa = load_a_frag(pl, 32, lane);

#pragma unroll
            for (int t = 0; t < 4; ++t) {
                v16bf bv = load_b_frag(&lV[buf][(t * 16) * 32], 32, lane);
                o[t] = wmma_bf16(pa, bv, o[t]);
            }
        }
        __syncthreads();                 // all waves done with buf before re-stage
    }

    int bb = bh / N_HEADS, h = bh % N_HEADS;
#pragma unroll
    for (int r = 0; r < 8; ++r) {
        float inv = 1.0f / lrow[r];
        int s = qb * 16 + r + 8 * halfL;
#pragma unroll
        for (int t = 0; t < 4; ++t) {
            int d = h * 64 + t * 16 + colL;
            O[((size_t)bb * SEQ + s) * D_MODEL + d] = f2bf(o[t][r] * inv);
        }
    }
}

extern "C" void kernel_launch(void* const* d_in, const int* in_sizes, int n_in,
                              void* d_out, int out_size, void* d_ws, size_t ws_size,
                              hipStream_t stream) {
    (void)in_sizes; (void)n_in; (void)out_size; (void)ws_size;
    const float* x  = (const float*)d_in[0];
    const float* Wq = (const float*)d_in[1];
    const float* bq = (const float*)d_in[2];
    const float* Wk = (const float*)d_in[3];
    const float* bk = (const float*)d_in[4];
    const float* Wv = (const float*)d_in[5];
    const float* bv = (const float*)d_in[6];
    const float* Wo = (const float*)d_in[7];
    const float* bo = (const float*)d_in[8];
    float* out = (float*)d_out;

    const size_t XE = (size_t)M_TOTAL * D_MODEL;       // 8,388,608 elems
    const size_t WE = (size_t)D_MODEL * D_MODEL;       // 1,048,576 elems
    char* ws = (char*)d_ws;
    size_t off = 0;
    auto carve = [&](size_t bytes) { void* p = ws + off; off += (bytes + 255) & ~(size_t)255; return p; };

    __bf16* xb   = (__bf16*)carve(XE * 2);
    __bf16* wqt  = (__bf16*)carve(WE * 2);
    __bf16* wkt  = (__bf16*)carve(WE * 2);
    __bf16* wvt  = (__bf16*)carve(WE * 2);
    __bf16* wot  = (__bf16*)carve(WE * 2);
    __bf16* qbuf = (__bf16*)carve(XE * 2);   // [B,H,S,Hd]
    __bf16* kbuf = (__bf16*)carve(XE * 2);   // [B,H,S,Hd]
    __bf16* vtb  = (__bf16*)carve(XE * 2);   // [B,H,Hd,S]
    __bf16* obuf = (__bf16*)carve(XE * 2);   // [B,S,D]

    dim3 blk(256);
    convert_x_kernel<<<(int)(XE / (256 * 8)), blk, 0, stream>>>(x, xb, (int)XE);
    transpose_w_kernel<<<(int)(WE / 256), blk, 0, stream>>>(Wq, wqt);
    transpose_w_kernel<<<(int)(WE / 256), blk, 0, stream>>>(Wk, wkt);
    transpose_w_kernel<<<(int)(WE / 256), blk, 0, stream>>>(Wv, wvt);
    transpose_w_kernel<<<(int)(WE / 256), blk, 0, stream>>>(Wo, wot);

    const int gemm_blocks = (M_TOTAL / 128) * (D_MODEL / 128);  // 512
    gemm_wmma_kernel<<<gemm_blocks, blk, 0, stream>>>(xb, wqt, bq, qbuf, 0);
    gemm_wmma_kernel<<<gemm_blocks, blk, 0, stream>>>(xb, wkt, bk, kbuf, 0);
    gemm_wmma_kernel<<<gemm_blocks, blk, 0, stream>>>(xb, wvt, bv, vtb,  1);

    const int fa_blocks = BATCH * N_HEADS * (SEQ / (16 * 8));   // 64 * 16 = 1024
    flash_attn_kernel<<<fa_blocks, blk, 0, stream>>>(qbuf, kbuf, vtb, obuf);

    gemm_wmma_kernel<<<gemm_blocks, blk, 0, stream>>>(obuf, wot, bo, out, 2);
}
